// DualNet_19353122636538
// MI455X (gfx1250) — compile-verified, matrix operand order checked by reading, not run on hardware
//
#include <hip/hip_runtime.h>
#include <hip/hip_bf16.h>
#include <stdint.h>

// ---------------- problem constants (from reference) ----------------
#define NN   20000
#define MP   20032    // NN padded to multiple of 64 (GEMM block-tile rows)
#define EE   160000
#define GG   100
#define FIN  1025
#define KP1  1056     // 1025 padded to multiple of 32
#define HH   512
#define EPS  1e-5f

typedef __attribute__((ext_vector_type(16))) _Float16 v16h;
typedef __attribute__((ext_vector_type(8)))  _Float16 v8h;
typedef __attribute__((ext_vector_type(8)))  float    v8f;

// ---------------- utility kernels ----------------
__global__ void k_zero_f(float* __restrict__ p, size_t n) {
    size_t i = (size_t)blockIdx.x * blockDim.x + threadIdx.x;
    size_t st = (size_t)gridDim.x * blockDim.x;
    for (; i < n; i += st) p[i] = 0.0f;
}
__global__ void k_zero_h(_Float16* __restrict__ p, size_t n) {
    size_t i = (size_t)blockIdx.x * blockDim.x + threadIdx.x;
    size_t st = (size_t)gridDim.x * blockDim.x;
    for (; i < n; i += st) p[i] = (_Float16)0.0f;
}
__global__ void k_zero_i(int* __restrict__ p, size_t n) {
    size_t i = (size_t)blockIdx.x * blockDim.x + threadIdx.x;
    size_t st = (size_t)gridDim.x * blockDim.x;
    for (; i < n; i += st) p[i] = 0;
}

// degree of src nodes
__global__ void k_deg(const int* __restrict__ src, float* __restrict__ deg, int e) {
    int i = blockIdx.x * blockDim.x + threadIdx.x;
    if (i < e) atomicAdd(&deg[src[i]], 1.0f);
}
__global__ void k_dis(const float* __restrict__ deg, float* __restrict__ dis, int n) {
    int i = blockIdx.x * blockDim.x + threadIdx.x;
    if (i < n) {
        float d = deg[i];
        dis[i] = (d > 0.0f) ? rsqrtf(fmaxf(d, 1.0f)) : 0.0f;
    }
}
__global__ void k_wedge(const int* __restrict__ src, const int* __restrict__ dst,
                        const float* __restrict__ dis, float* __restrict__ we, int e) {
    int i = blockIdx.x * blockDim.x + threadIdx.x;
    if (i < e) we[i] = -dis[src[i]] * dis[dst[i]];
}

// f32 [rowsReal,K] -> f16 [rowsPad,Kp] (zero padded in both dims)
__global__ void k_cast_pad(const float* __restrict__ X, _Float16* __restrict__ Y,
                           int rowsReal, int rowsPad, int K, int Kp) {
    size_t tot = (size_t)rowsPad * Kp;
    size_t i = (size_t)blockIdx.x * blockDim.x + threadIdx.x;
    size_t st = (size_t)gridDim.x * blockDim.x;
    for (; i < tot; i += st) {
        int k = (int)(i % Kp);
        size_t r = i / Kp;
        Y[i] = (k < K && r < (size_t)rowsReal) ? (_Float16)X[r * (size_t)K + k]
                                               : (_Float16)0.0f;
    }
}
// W [K,NC] f32 -> Wt [NC,Kp] f16 (transposed, zero padded)
__global__ void k_wtrans(const float* __restrict__ W, _Float16* __restrict__ Wt,
                         int K, int NC, int Kp) {
    size_t tot = (size_t)NC * Kp;
    size_t i = (size_t)blockIdx.x * blockDim.x + threadIdx.x;
    size_t st = (size_t)gridDim.x * blockDim.x;
    for (; i < tot; i += st) {
        int k = (int)(i % Kp);
        int c = (int)(i / Kp);
        Wt[i] = (k < K) ? (_Float16)W[(size_t)k * NC + c] : (_Float16)0.0f;
    }
}

// ---------------- WMMA GEMM ----------------
// C[M,NC] = A[M,K]f16 * Bt[NC,K]f16 (+bias).  M%64==0, NC%128==0, K%32==0.
// block = 256 threads = 8 waves (2 row-groups x 4 col-groups); wave tile 32x32
// via 2x2 register blocking (4 accumulators, 4 WMMAs per K-step of 32).
// f16 A-fragment (16x32): lanes 0-15 row=lane, K 0-7 & 16-23; lanes 16-31 same
// rows, K 8-15 & 24-31.  B fragment mirrors A with Bt row = output column.
// C/D: VGPR p -> row p (lanes 0-15) / p+8 (lanes 16-31), col = lane&15.
__device__ inline v16h ldfrag(const _Float16* p) {
    v8h lo = *(const v8h*)(p);
    v8h hv = *(const v8h*)(p + 16);
    v16h r;
#pragma unroll
    for (int i = 0; i < 8; ++i) { r[i] = lo[i]; r[i + 8] = hv[i]; }
    return r;
}

__global__ void __launch_bounds__(256)
k_gemm_f16(const _Float16* __restrict__ A, const _Float16* __restrict__ Bt,
           const float* __restrict__ bias, float* __restrict__ C,
           int M, int K, int NC) {
    const int lane = threadIdx.x & 31;
    const int wid  = threadIdx.x >> 5;                 // 0..7
    const int rowBase = blockIdx.x * 64  + (wid >> 2) * 32;
    const int colBase = blockIdx.y * 128 + (wid & 3) * 32;
    const int hi   = lane >> 4;                        // 0 or 1
    const int l15  = lane & 15;

    const _Float16* ap0 = A  + (size_t)(rowBase + l15) * K      + hi * 8;
    const _Float16* ap1 = A  + (size_t)(rowBase + 16 + l15) * K + hi * 8;
    const _Float16* bp0 = Bt + (size_t)(colBase + l15) * K      + hi * 8;
    const _Float16* bp1 = Bt + (size_t)(colBase + 16 + l15) * K + hi * 8;

    v8f acc00 = {}, acc01 = {}, acc10 = {}, acc11 = {};
    for (int k0 = 0; k0 < K; k0 += 32) {
        if (k0 + 128 < K) {
            __builtin_prefetch(ap0 + k0 + 128, 0, 1);
            __builtin_prefetch(ap1 + k0 + 128, 0, 1);
            __builtin_prefetch(bp0 + k0 + 128, 0, 1);
            __builtin_prefetch(bp1 + k0 + 128, 0, 1);
        }
        v16h a0 = ldfrag(ap0 + k0);
        v16h a1 = ldfrag(ap1 + k0);
        v16h b0 = ldfrag(bp0 + k0);
        v16h b1 = ldfrag(bp1 + k0);
        acc00 = __builtin_amdgcn_wmma_f32_16x16x32_f16(false, a0, false, b0,
                                                       (short)0, acc00, false, false);
        acc01 = __builtin_amdgcn_wmma_f32_16x16x32_f16(false, a0, false, b1,
                                                       (short)0, acc01, false, false);
        acc10 = __builtin_amdgcn_wmma_f32_16x16x32_f16(false, a1, false, b0,
                                                       (short)0, acc10, false, false);
        acc11 = __builtin_amdgcn_wmma_f32_16x16x32_f16(false, a1, false, b1,
                                                       (short)0, acc11, false, false);
    }
    const int c0 = colBase + l15;
    const int c1 = c0 + 16;
    const float bv0 = bias ? bias[c0] : 0.0f;
    const float bv1 = bias ? bias[c1] : 0.0f;
#pragma unroll
    for (int p = 0; p < 8; ++p) {
        int r0 = rowBase + p + hi * 8;
        int r1 = r0 + 16;
        C[(size_t)r0 * NC + c0] = acc00[p] + bv0;
        C[(size_t)r0 * NC + c1] = acc01[p] + bv1;
        C[(size_t)r1 * NC + c0] = acc10[p] + bv0;
        C[(size_t)r1 * NC + c1] = acc11[p] + bv1;
    }
}

// edge aggregation: Acc[dst] += w_edge * Y1[src]   (one block per edge)
__global__ void k_edge_scatter(const int* __restrict__ src, const int* __restrict__ dst,
                               const float* __restrict__ we,
                               const float* __restrict__ Y1, float* __restrict__ Acc,
                               int cols) {
    int e = blockIdx.x;
    int s = src[e], d = dst[e];
    float w = we[e];
    const float* y = Y1 + (size_t)s * cols;
    float* a = Acc + (size_t)d * cols;
    for (int c = threadIdx.x; c < cols; c += blockDim.x)
        atomicAdd(&a[c], w * y[c]);
}

// ---------------- BatchNorm (training-mode batch stats, biased var) ----------------
__global__ void k_bn_stats(const float* __restrict__ X, int rows, int cols,
                           const float* __restrict__ g, const float* __restrict__ b,
                           float* __restrict__ scale, float* __restrict__ shift) {
    int c = blockIdx.x;
    __shared__ float ssum[256], ssq[256];
    float s = 0.0f, q = 0.0f;
    for (int r = threadIdx.x; r < rows; r += blockDim.x) {
        float v = X[(size_t)r * cols + c];
        s += v; q += v * v;
    }
    ssum[threadIdx.x] = s; ssq[threadIdx.x] = q;
    __syncthreads();
    for (int st = 128; st > 0; st >>= 1) {
        if (threadIdx.x < st) {
            ssum[threadIdx.x] += ssum[threadIdx.x + st];
            ssq[threadIdx.x]  += ssq[threadIdx.x + st];
        }
        __syncthreads();
    }
    if (threadIdx.x == 0) {
        float m = ssum[0] / rows;
        float var = ssq[0] / rows - m * m;
        float rstd = rsqrtf(var + EPS);
        float sc = g[c] * rstd;
        scale[c] = sc;
        shift[c] = b[c] - m * sc;
    }
}
// y = relu(x*scale + shift) -> optional f32 and/or f16 outputs
__global__ void k_bn_apply(const float* __restrict__ X,
                           const float* __restrict__ scale, const float* __restrict__ shift,
                           float* __restrict__ outF, _Float16* __restrict__ outH,
                           size_t rows, int cols) {
    size_t tot = rows * cols;
    size_t i = (size_t)blockIdx.x * blockDim.x + threadIdx.x;
    size_t st = (size_t)gridDim.x * blockDim.x;
    for (; i < tot; i += st) {
        int c = (int)(i % cols);
        float v = fmaxf(X[i] * scale[c] + shift[c], 0.0f);
        if (outF) outF[i] = v;
        if (outH) outH[i] = (_Float16)v;
    }
}

// ---------------- graph pooling (batch is sorted) ----------------
__global__ void k_segbounds(const int* __restrict__ batch, int* __restrict__ gs,
                            int* __restrict__ ge, int n) {
    int i = blockIdx.x * blockDim.x + threadIdx.x;
    if (i >= n) return;
    int b = batch[i];
    if (i == 0 || batch[i - 1] != b) gs[b] = i;
    if (i == n - 1 || batch[i + 1] != b) ge[b] = i + 1;
}
__global__ void k_pool(const float* __restrict__ H2, const int* __restrict__ gs,
                       const int* __restrict__ ge, float* __restrict__ feat0, int cols) {
    int g = blockIdx.x;
    int c = blockIdx.y * blockDim.x + threadIdx.x;
    int s = gs[g], e = ge[g];
    float mx = -3.4e38f, sum = 0.0f;
    for (int i = s; i < e; ++i) {
        float v = H2[(size_t)i * cols + c];
        sum += v;
        mx = fmaxf(mx, v);
    }
    int cnt = e - s;
    if (cnt == 0) mx = 0.0f;
    feat0[(size_t)g * (2 * cols) + c] = mx;                                   // gmax
    feat0[(size_t)g * (2 * cols) + cols + c] = sum / fmaxf((float)cnt, 1.0f); // gavg
}

// ---------------- small dense layers (heads) ----------------
__global__ void k_linear(const float* __restrict__ A, const float* __restrict__ W,
                         const float* __restrict__ b, float* __restrict__ C,
                         int M, int K, int NC) {
    int idx = blockIdx.x * blockDim.x + threadIdx.x;
    if (idx >= M * NC) return;
    int m = idx / NC, n = idx % NC;
    float acc = b[n];
    const float* a = A + (size_t)m * K;
    for (int k = 0; k < K; ++k) acc += a[k] * W[(size_t)k * NC + n];
    C[idx] = acc;
}
// final layer: relu(linear) then softmax; one thread per row, NC <= 8
__global__ void k_head(const float* __restrict__ A, const float* __restrict__ W,
                       const float* __restrict__ b, float* __restrict__ out,
                       int M, int K, int NC) {
    int m = blockIdx.x * blockDim.x + threadIdx.x;
    if (m >= M) return;
    float t[8];
    float mx = -3.4e38f;
    for (int n = 0; n < NC; ++n) {
        float acc = b[n];
        const float* a = A + (size_t)m * K;
        for (int k = 0; k < K; ++k) acc += a[k] * W[(size_t)k * NC + n];
        acc = fmaxf(acc, 0.0f);
        t[n] = acc;
        mx = fmaxf(mx, acc);
    }
    float den = 0.0f;
    for (int n = 0; n < NC; ++n) { t[n] = expf(t[n] - mx); den += t[n]; }
    for (int n = 0; n < NC; ++n) out[(size_t)m * NC + n] = t[n] / den;
}

// ---------------- host driver ----------------
extern "C" void kernel_launch(void* const* d_in, const int* in_sizes, int n_in,
                              void* d_out, int out_size, void* d_ws, size_t ws_size,
                              hipStream_t stream) {
    (void)in_sizes; (void)n_in; (void)out_size; (void)ws_size;

    const float* x       = (const float*)d_in[0];   // [N,1025]
    const int*   src     = (const int*)d_in[1];     // edge_index[0]
    const int*   dst     = (const int*)d_in[1] + EE;// edge_index[1]
    const int*   batch   = (const int*)d_in[2];     // [N]
    // d_in[3] feature_s, d_in[4] flag: unused (flag==2 path)
    const float* conv1_w = (const float*)d_in[5];   // [2,1025,512]
    const float* conv1_b = (const float*)d_in[6];
    const float* conv2_w = (const float*)d_in[7];   // [2,512,512]
    const float* conv2_b = (const float*)d_in[8];
    const float* bn1_g = (const float*)d_in[9],  *bn1_b = (const float*)d_in[10];
    const float* bn2_g = (const float*)d_in[11], *bn2_b = (const float*)d_in[12];
    const float* lin01_w = (const float*)d_in[13], *lin01_b = (const float*)d_in[14];
    const float* lin02_w = (const float*)d_in[15], *lin02_b = (const float*)d_in[16];
    const float* lin03_w = (const float*)d_in[17], *lin03_b = (const float*)d_in[18];
    const float* bn01_g = (const float*)d_in[19], *bn01_b = (const float*)d_in[20];
    const float* bn02_g = (const float*)d_in[21], *bn02_b = (const float*)d_in[22];
    const float* lin11_w = (const float*)d_in[23], *lin11_b = (const float*)d_in[24];
    const float* lin12_w = (const float*)d_in[25], *lin12_b = (const float*)d_in[26];
    const float* lin13_w = (const float*)d_in[27], *lin13_b = (const float*)d_in[28];
    const float* bn11_g = (const float*)d_in[29], *bn11_b = (const float*)d_in[30];
    const float* bn12_g = (const float*)d_in[31], *bn12_b = (const float*)d_in[32];

    // output layout: output0[200] | output1[500] | feature[N*H] | feature0[G*1024] | feature1[G*256]
    float* out0  = (float*)d_out;
    float* out1  = out0 + 200;
    float* feat  = out1 + 500;
    float* feat0 = feat + (size_t)NN * HH;
    float* feat1 = feat0 + (size_t)GG * 1024;

    // workspace carving (256B aligned); padded-row buffers for the WMMA GEMMs
    uintptr_t cur = (uintptr_t)d_ws;
    auto carve = [&](size_t bytes) -> void* {
        void* p = (void*)cur;
        cur += (bytes + 255) & ~(size_t)255;
        return p;
    };
    float*    deg   = (float*)carve((size_t)NN * 4);
    float*    dis   = (float*)carve((size_t)NN * 4);
    float*    we    = (float*)carve((size_t)EE * 4);
    _Float16* xh    = (_Float16*)carve((size_t)MP * KP1 * 2);
    _Float16* w0t   = (_Float16*)carve((size_t)HH * KP1 * 2);
    _Float16* w1t   = (_Float16*)carve((size_t)HH * KP1 * 2);
    float*    y0    = (float*)carve((size_t)MP * HH * 4);
    float*    y1    = (float*)carve((size_t)MP * HH * 4);
    _Float16* h1h   = (_Float16*)carve((size_t)MP * HH * 2);
    _Float16* w0t2  = (_Float16*)carve((size_t)HH * HH * 2);
    _Float16* w1t2  = (_Float16*)carve((size_t)HH * HH * 2);
    float*    scale = (float*)carve(HH * 4);
    float*    shift = (float*)carve(HH * 4);
    int*      gs    = (int*)carve(GG * 4);
    int*      ge    = (int*)carve(GG * 4);
    float*    x0a   = (float*)carve((size_t)GG * 512 * 4);
    float*    x0b   = (float*)carve((size_t)GG * 256 * 4);
    float*    x1a   = (float*)carve((size_t)GG * 512 * 4);
    float*    x1b   = (float*)carve((size_t)GG * 256 * 4);

    // --- edge weights: deg -> dis -> w_edge ---
    k_zero_f<<<128, 256, 0, stream>>>(deg, (size_t)NN);
    k_deg<<<(EE + 255) / 256, 256, 0, stream>>>(src, deg, EE);
    k_dis<<<(NN + 255) / 256, 256, 0, stream>>>(deg, dis, NN);
    k_wedge<<<(EE + 255) / 256, 256, 0, stream>>>(src, dst, dis, we, EE);

    // --- conv1: y0 = x@W0 + b ; y1 = x@W1 ; y0 += scatter(w * y1[src] -> dst) ---
    k_cast_pad<<<4096, 256, 0, stream>>>(x, xh, NN, MP, FIN, KP1);
    k_wtrans<<<2048, 256, 0, stream>>>(conv1_w, w0t, FIN, HH, KP1);
    k_wtrans<<<2048, 256, 0, stream>>>(conv1_w + (size_t)FIN * HH, w1t, FIN, HH, KP1);
    {
        dim3 g(MP / 64, HH / 128);
        k_gemm_f16<<<g, 256, 0, stream>>>(xh, w0t, conv1_b, y0, MP, KP1, HH);
        k_gemm_f16<<<g, 256, 0, stream>>>(xh, w1t, nullptr, y1, MP, KP1, HH);
    }
    k_edge_scatter<<<EE, 256, 0, stream>>>(src, dst, we, y1, y0, HH);
    k_bn_stats<<<HH, 256, 0, stream>>>(y0, NN, HH, bn1_g, bn1_b, scale, shift);
    k_bn_apply<<<8192, 256, 0, stream>>>(y0, scale, shift, nullptr, h1h, (size_t)NN, HH);
    k_zero_h<<<32, 256, 0, stream>>>(h1h + (size_t)NN * HH, (size_t)(MP - NN) * HH); // pad rows

    // --- conv2 (input h1 as f16) ---
    k_wtrans<<<1024, 256, 0, stream>>>(conv2_w, w0t2, HH, HH, HH);
    k_wtrans<<<1024, 256, 0, stream>>>(conv2_w + (size_t)HH * HH, w1t2, HH, HH, HH);
    {
        dim3 g(MP / 64, HH / 128);
        k_gemm_f16<<<g, 256, 0, stream>>>(h1h, w0t2, conv2_b, y0, MP, HH, HH);
        k_gemm_f16<<<g, 256, 0, stream>>>(h1h, w1t2, nullptr, y1, MP, HH, HH);
    }
    k_edge_scatter<<<EE, 256, 0, stream>>>(src, dst, we, y1, y0, HH);
    k_bn_stats<<<HH, 256, 0, stream>>>(y0, NN, HH, bn2_g, bn2_b, scale, shift);
    k_bn_apply<<<8192, 256, 0, stream>>>(y0, scale, shift, feat, nullptr, (size_t)NN, HH); // feature

    // --- pooling (batch sorted) -> feature0 = [gmax | gavg] [G,1024] ---
    k_zero_i<<<1, 256, 0, stream>>>(gs, GG);
    k_zero_i<<<1, 256, 0, stream>>>(ge, GG);
    k_segbounds<<<(NN + 255) / 256, 256, 0, stream>>>(batch, gs, ge, NN);
    {
        dim3 g(GG, HH / 256);
        k_pool<<<g, 256, 0, stream>>>(feat, gs, ge, feat0, HH);
    }

    // --- head 0 ---
    k_linear<<<(GG * 512 + 255) / 256, 256, 0, stream>>>(feat0, lin01_w, lin01_b, x0a, GG, 1024, 512);
    k_bn_stats<<<512, 256, 0, stream>>>(x0a, GG, 512, bn01_g, bn01_b, scale, shift);
    k_bn_apply<<<256, 256, 0, stream>>>(x0a, scale, shift, x0a, nullptr, (size_t)GG, 512);
    k_linear<<<(GG * 256 + 255) / 256, 256, 0, stream>>>(x0a, lin02_w, lin02_b, x0b, GG, 512, 256);
    k_bn_stats<<<256, 256, 0, stream>>>(x0b, GG, 256, bn02_g, bn02_b, scale, shift);
    k_bn_apply<<<128, 256, 0, stream>>>(x0b, scale, shift, x0b, nullptr, (size_t)GG, 256);
    k_head<<<1, 128, 0, stream>>>(x0b, lin03_w, lin03_b, out0, GG, 256, 2);

    // --- head 1 ---
    k_linear<<<(GG * 512 + 255) / 256, 256, 0, stream>>>(feat0, lin11_w, lin11_b, x1a, GG, 1024, 512);
    k_bn_stats<<<512, 256, 0, stream>>>(x1a, GG, 512, bn11_g, bn11_b, scale, shift);
    k_bn_apply<<<256, 256, 0, stream>>>(x1a, scale, shift, x1a, nullptr, (size_t)GG, 512);
    k_linear<<<(GG * 256 + 255) / 256, 256, 0, stream>>>(x1a, lin12_w, lin12_b, x1b, GG, 512, 256);
    k_bn_stats<<<256, 256, 0, stream>>>(x1b, GG, 256, bn12_g, bn12_b, scale, shift);
    k_bn_apply<<<128, 256, 0, stream>>>(x1b, scale, shift, feat1, nullptr, (size_t)GG, 256); // feature1
    k_head<<<1, 128, 0, stream>>>(feat1, lin13_w, lin13_b, out1, GG, 256, 5);
}
